// DualTagger_65532611002431
// MI455X (gfx1250) — matrix operand order repaired and checked by reading, not compiled, not run on hardware
//
#include <hip/hip_runtime.h>
#include <hip/hip_bf16.h>

// ---------------------------------------------------------------------------
// DualTagger (char-LSTM + word-LSTM tagger) for MI455X / gfx1250.
//
//   Phase A (parallel):
//     1) char gate table: gtab[c][g] = Wih_c@char_emb[c] + b_c   (CV=128 only!)
//     2) word input gates: xg_w = we@Wih_w[:, :D].T + b_w  via WMMA f16 GEMM
//     3) pack [Wih_w[:,D:] | Whh_w] into WMMA A-fragment order (f16)
//   Phase B (serial, 1 workgroup of 512 threads = 16 wave32):
//     recurrence with v_wmma_f32_16x16x32_f16 matvecs; char recurrent weights
//     resident in VGPRs; independent accumulators to avoid WMMA RAW bubbles;
//     f16 shadow state in LDS so B fragments are single 32B vector loads.
// ---------------------------------------------------------------------------

typedef __attribute__((ext_vector_type(16))) _Float16 v16h;
typedef __attribute__((ext_vector_type(8)))  float    v8f;

#define S_   8192
#define LC_  16
#define D_   256
#define DC_  64
#define HC_  128
#define H_   256
#define T_   17
#define GC_  (4 * HC_)      // 512 char gates
#define GW_  (4 * H_)       // 1024 word gates
#define KW_  (HC_ + H_)     // 384 recurrent input dim of word LSTM ([hc;hw])
#define NTILE_W (GW_ / 16)  // 64 word-gate row tiles
#define NKC_W   (KW_ / 32)  // 12 K-chunks of 32 for word matvec

// A-matrix (16x32 f16) K index for (lane, half-element hv): K = hv + 16*(lane>=16), matching B
__device__ __forceinline__ int a_kmap(int lane, int hv) {
    return hv + ((lane >= 16) ? 16 : 0);   // 0..31
}
// B-matrix (32x16 f16): lane = column N (mod 16), K = hv + 16*(lane>=16)
__device__ __forceinline__ int b_kmap(int lane, int hv) {
    return hv + ((lane >= 16) ? 16 : 0);
}

__device__ __forceinline__ float sigmoidf_(float x) {
    return 1.0f / (1.0f + __expf(-x));
}

// ---------------------------------------------------------------------------
// Kernel 1: char gate lookup table  gtab[CV_][GC_]
// ---------------------------------------------------------------------------
__global__ void k_gtab(const float* __restrict__ char_emb,
                       const float* __restrict__ Wih_c,
                       const float* __restrict__ b_c,
                       float* __restrict__ gtab) {
    int c = blockIdx.x;        // 0..127
    int g = threadIdx.x;       // 0..511
    float acc = b_c[g];
    #pragma unroll 8
    for (int d = 0; d < DC_; ++d)
        acc = fmaf(char_emb[c * DC_ + d], Wih_c[g * DC_ + d], acc);
    gtab[c * GC_ + g] = acc;
}

// ---------------------------------------------------------------------------
// Kernel 2: pack word recurrent weights [Wih_w[:,D:] | Whh_w] (1024x384) into
// WMMA A-fragment order:  wpack[((tile*12 + kc)*32 + lane)*16 + hv]
// ---------------------------------------------------------------------------
__global__ void k_wpack(const float* __restrict__ Wih_w,
                        const float* __restrict__ Whh_w,
                        _Float16* __restrict__ wpack) {
    int idx = blockIdx.x * 256 + threadIdx.x;
    const int total = NTILE_W * NKC_W * 32 * 16;   // 393216
    if (idx >= total) return;
    int hv   = idx & 15;
    int lane = (idx >> 4) & 31;
    int kc   = (idx >> 9) % NKC_W;
    int tile = (idx >> 9) / NKC_W;
    int row  = tile * 16 + (lane & 15);
    int K    = kc * 32 + a_kmap(lane, hv);
    float v  = (K < HC_) ? Wih_w[row * KW_ + D_ + K]
                         : Whh_w[row * H_ + (K - HC_)];
    wpack[idx] = (_Float16)v;
}

// ---------------------------------------------------------------------------
// Kernel 3: xg_w[s][g] = b_w[g] + sum_d we[sentence[s]][d] * Wih_w[g][d]
// One wave32 per 16x16 tile; K=256 in 8 chunks of 32, f16 WMMA, f32 accum.
// Two independent accumulators to avoid WMMA->WMMA RAW bubbles.
// ---------------------------------------------------------------------------
__global__ __launch_bounds__(32)
void k_xgw(const int* __restrict__ sentence,
           const float* __restrict__ word_emb,
           const float* __restrict__ Wih_w,
           const float* __restrict__ b_w,
           float* __restrict__ xg_w) {
    int m0 = blockIdx.x * 16;              // word tile
    int n0 = blockIdx.y * 16;              // gate tile
    int lane = threadIdx.x;
    int lm = lane & 15;
    bool hi = lane >= 16;

    int wid = sentence[m0 + lm];
    const float* arow = word_emb + (long)wid * D_;
    const float* brow = Wih_w + (long)(n0 + lm) * KW_;

    v8f c0 = {}, c1 = {};
    float bb = b_w[n0 + lm];
    #pragma unroll
    for (int r = 0; r < 8; ++r) c0[r] = bb;

    #pragma unroll
    for (int kc = 0; kc < 8; ++kc) {
        v16h a, b;
        #pragma unroll
        for (int j = 0; j < 16; ++j) {
            a[j] = (_Float16)arow[kc * 32 + a_kmap(lane, j)];
            b[j] = (_Float16)brow[kc * 32 + b_kmap(lane, j)];
        }
        if (kc & 1)
            c1 = __builtin_amdgcn_wmma_f32_16x16x32_f16(false, a, false, b,
                                                        (short)0, c1, false, false);
        else
            c0 = __builtin_amdgcn_wmma_f32_16x16x32_f16(false, a, false, b,
                                                        (short)0, c0, false, false);
    }
    v8f c = c0 + c1;
    #pragma unroll
    for (int r = 0; r < 8; ++r) {
        int m = m0 + r + (hi ? 8 : 0);
        xg_w[(long)m * GW_ + n0 + lm] = c[r];
    }
}

// ---------------------------------------------------------------------------
// Kernel 4: the serial recurrence. One block, 512 threads (16 wave32).
// Wave w owns char-gate rows [32w,32w+32) and word-gate tiles [4w,4w+4).
// ---------------------------------------------------------------------------
__global__ __launch_bounds__(512)
void k_seq(const int* __restrict__ word_chars,
           const int* __restrict__ char_lens,
           const float* __restrict__ Whh_c,
           const float* __restrict__ gtab,
           const float* __restrict__ xg_w,
           const _Float16* __restrict__ wpack,
           const float* __restrict__ Wout,
           const float* __restrict__ bout,
           float* __restrict__ out) {
    __shared__ float sh_cc[HC_];                       // char cell state (f32)
    __shared__ float sh_hw[H_], sh_cw[H_];             // word states (f32)
    __shared__ __align__(32) _Float16 sh_hc_h[HC_];    // f16 shadow of hc
    __shared__ __align__(32) _Float16 sh_hw_h[H_];     // f16 shadow of hw
    __shared__ __align__(32) float sh_gc[GC_];
    __shared__ __align__(32) float sh_gw[GW_];
    __shared__ int sh_len;

    const int tid  = threadIdx.x;
    const int lane = tid & 31;
    const int w    = tid >> 5;      // wave 0..15
    const int lm   = lane & 15;
    const bool hi  = lane >= 16;

    if (tid < HC_) { sh_hc_h[tid] = (_Float16)0.f; sh_cc[tid] = 0.f; }
    if (tid < H_)  { sh_hw[tid] = 0.f; sh_cw[tid] = 0.f; sh_hw_h[tid] = (_Float16)0.f; }

    // Char recurrent weights resident as WMMA A fragments:
    // 2 row-tiles x 4 K-chunks of 32 -> 8 fragments / wave (64 VGPRs).
    v16h ac[2][4];
    #pragma unroll
    for (int mt = 0; mt < 2; ++mt) {
        int row = w * 32 + mt * 16 + lm;
        #pragma unroll
        for (int kc = 0; kc < 4; ++kc) {
            v16h f;
            #pragma unroll
            for (int j = 0; j < 16; ++j)
                f[j] = (_Float16)Whh_c[row * HC_ + kc * 32 + a_kmap(lane, j)];
            ac[mt][kc] = f;
        }
    }
    __syncthreads();

    for (int s = 0; s < S_; ++s) {
        if (tid == 0) {
            sh_len = char_lens[s];
            if (s + 1 < S_) __builtin_prefetch(xg_w + (long)(s + 1) * GW_, 0, 0);
        }
        __syncthreads();

        // ---- 16 char-LSTM steps (state chains across words) ----
        for (int t = 0; t < LC_; ++t) {
            // B fragments: column N=0 carries hc; lanes 0/16 each load 32B.
            v16h bf[4];
            #pragma unroll
            for (int kc = 0; kc < 4; ++kc) {
                v16h f = {};
                if (lm == 0)
                    f = *(const v16h*)(sh_hc_h + kc * 32 + (hi ? 16 : 0));
                bf[kc] = f;
            }
            int cid = word_chars[s * LC_ + t];
            const float* gt = gtab + cid * GC_;

            #pragma unroll
            for (int mt = 0; mt < 2; ++mt) {
                int rowbase = w * 32 + mt * 16;
                // 4 independent accumulators: no WMMA->WMMA RAW chain.
                v8f c0 = {}, c1 = {}, c2 = {}, c3 = {};
                if (lm == 0)
                    c0 = *(const v8f*)(gt + rowbase + (hi ? 8 : 0));
                c0 = __builtin_amdgcn_wmma_f32_16x16x32_f16(
                        false, ac[mt][0], false, bf[0], (short)0, c0, false, false);
                c1 = __builtin_amdgcn_wmma_f32_16x16x32_f16(
                        false, ac[mt][1], false, bf[1], (short)0, c1, false, false);
                c2 = __builtin_amdgcn_wmma_f32_16x16x32_f16(
                        false, ac[mt][2], false, bf[2], (short)0, c2, false, false);
                c3 = __builtin_amdgcn_wmma_f32_16x16x32_f16(
                        false, ac[mt][3], false, bf[3], (short)0, c3, false, false);
                v8f csum = (c0 + c1) + (c2 + c3);
                if (lm == 0)
                    *(v8f*)(sh_gc + rowbase + (hi ? 8 : 0)) = csum;
            }
            __syncthreads();

            if (tid < HC_ && t < sh_len) {   // freeze state past ragged length
                float gi = sh_gc[tid];
                float gf = sh_gc[HC_ + tid];
                float gg = sh_gc[2 * HC_ + tid];
                float go = sh_gc[3 * HC_ + tid];
                float c2s = sigmoidf_(gf) * sh_cc[tid]
                          + sigmoidf_(gi) * tanhf(gg);
                sh_cc[tid] = c2s;
                sh_hc_h[tid] = (_Float16)(sigmoidf_(go) * tanhf(c2s));
            }
            __syncthreads();
        }

        // ---- word-LSTM step: gates = xg_w[s] + Wrec(1024x384) @ [hc;hw] ----
        // B fragments: z = [hc(128); hw(256)] in f16, chunk kc of 32.
        v16h bw[NKC_W];
        #pragma unroll
        for (int kc = 0; kc < NKC_W; ++kc) {
            v16h f = {};
            if (lm == 0) {
                const _Float16* zp = (kc < 4) ? (sh_hc_h + kc * 32)
                                              : (sh_hw_h + (kc - 4) * 32);
                f = *(const v16h*)(zp + (hi ? 16 : 0));
            }
            bw[kc] = f;
        }

        const float* xw = xg_w + (long)s * GW_;
        #pragma unroll
        for (int mt = 0; mt < 4; ++mt) {
            int tile = w * 4 + mt;
            int rowbase = tile * 16;
            // 4 round-robin accumulators: chains of depth 3, fully overlapped.
            v8f acc[4] = {{}, {}, {}, {}};
            if (lm == 0)
                acc[0] = *(const v8f*)(xw + rowbase + (hi ? 8 : 0));
            const _Float16* ap = wpack + (long)tile * NKC_W * 32 * 16;
            #pragma unroll
            for (int kc = 0; kc < NKC_W; ++kc) {
                v16h a = *(const v16h*)(ap + (kc * 32 + lane) * 16);
                acc[kc & 3] = __builtin_amdgcn_wmma_f32_16x16x32_f16(
                        false, a, false, bw[kc], (short)0, acc[kc & 3], false, false);
            }
            v8f csum = (acc[0] + acc[1]) + (acc[2] + acc[3]);
            if (lm == 0)
                *(v8f*)(sh_gw + rowbase + (hi ? 8 : 0)) = csum;
        }
        __syncthreads();

        if (tid < H_) {
            float gi = sh_gw[tid];
            float gf = sh_gw[H_ + tid];
            float gg = sh_gw[2 * H_ + tid];
            float go = sh_gw[3 * H_ + tid];
            float c2s = sigmoidf_(gf) * sh_cw[tid] + sigmoidf_(gi) * tanhf(gg);
            float h2 = sigmoidf_(go) * tanhf(c2s);
            sh_cw[tid] = c2s;
            sh_hw[tid] = h2;
            sh_hw_h[tid] = (_Float16)h2;
        }
        __syncthreads();

        // ---- logits: Wout(17x256) @ hw + bout ----
        if (tid < T_) {
            float acc = bout[tid];
            #pragma unroll 8
            for (int d = 0; d < H_; ++d)
                acc = fmaf(Wout[tid * H_ + d], sh_hw[d], acc);
            out[(long)s * T_ + tid] = acc;
        }
        __syncthreads();
    }
}

// ---------------------------------------------------------------------------
extern "C" void kernel_launch(void* const* d_in, const int* in_sizes, int n_in,
                              void* d_out, int out_size, void* d_ws, size_t ws_size,
                              hipStream_t stream) {
    const int*   sentence   = (const int*)d_in[0];
    const int*   word_chars = (const int*)d_in[1];
    const int*   char_lens  = (const int*)d_in[2];
    const float* word_emb   = (const float*)d_in[3];
    const float* char_emb   = (const float*)d_in[4];
    const float* Wih_c      = (const float*)d_in[5];
    const float* Whh_c      = (const float*)d_in[6];
    const float* b_c        = (const float*)d_in[7];
    const float* Wih_w      = (const float*)d_in[8];
    const float* Whh_w      = (const float*)d_in[9];
    const float* b_w        = (const float*)d_in[10];
    const float* Wout       = (const float*)d_in[11];
    const float* bout       = (const float*)d_in[12];
    float* out = (float*)d_out;

    char* ws = (char*)d_ws;
    float*    xg_w  = (float*)ws;                                   // 32 MB
    float*    gtab  = (float*)(ws + (size_t)S_ * GW_ * 4);          // 256 KB
    _Float16* wpack = (_Float16*)(ws + (size_t)S_ * GW_ * 4
                                     + (size_t)128 * GC_ * 4);      // 768 KB

    k_gtab<<<128, GC_, 0, stream>>>(char_emb, Wih_c, b_c, gtab);
    {
        const int total = NTILE_W * NKC_W * 32 * 16;
        k_wpack<<<(total + 255) / 256, 256, 0, stream>>>(Wih_w, Whh_w, wpack);
    }
    k_xgw<<<dim3(S_ / 16, GW_ / 16), 32, 0, stream>>>(sentence, word_emb,
                                                      Wih_w, b_w, xg_w);
    k_seq<<<1, 512, 0, stream>>>(word_chars, char_lens, Whh_c, gtab, xg_w,
                                 wpack, Wout, bout, out);
}
